// ToMeBertAttention_18107582120760
// MI455X (gfx1250) — matrix operand order repaired and checked by reading, not compiled, not run on hardware
//
#include <hip/hip_runtime.h>

// ---------------- problem constants ----------------
#define BB   4
#define TT   2048
#define HH   12
#define DD   64
#define HID  768
#define RR   512
#define T2   1536
#define SCALE 0.125f   // 1/sqrt(64)

// ---------------- WMMA types ----------------
typedef __attribute__((ext_vector_type(16))) __bf16 v16bf;
typedef __attribute__((ext_vector_type(8)))  float  v8f;

// ---------------- bf16 helpers ----------------
__device__ __forceinline__ unsigned short f32_to_bf16_bits(float f) {
    union { float f; unsigned u; } v; v.f = f;
    unsigned u = v.u;
    unsigned r = u + 0x7FFFu + ((u >> 16) & 1u);   // round-to-nearest-even
    return (unsigned short)(r >> 16);
}
__device__ __forceinline__ __bf16 us_to_bf(unsigned short s) {
    union { unsigned short s; __bf16 h; } v; v.s = s; return v.h;
}

// ---------------- vectorized WMMA fragment loaders ----------------
// A-matrix 16x32 bf16 from row-major [M][K] source.
// Lane layout (ISA): m = lane&15, half = lane>>4.
//   VGPR i<4 : K = 2i + 8*half      -> 8 contiguous elements at K = 8*half
//   VGPR i>=4: K = 16+2(i-4)+8*half -> 8 contiguous elements at K = 16+8*half
// => two 16-byte loads per lane.
__device__ __forceinline__ v16bf load_A_vec(const unsigned short* p, int ld, int lane) {
    int m = lane & 15, half = lane >> 4;
    const uint4* q = (const uint4*)(p + (size_t)m * ld + half * 8);
    union { v16bf v; uint4 q[2]; } r;
    r.q[0] = q[0];        // elements 0..7
    r.q[1] = q[2];        // elements 16..23 (+32 bytes)
    return r.v;
}

// B-matrix 32x16 bf16 where B[k][n] = src[n*ld + dbase + k]  (i.e. the source
// holds the n-dimension as rows: transposed weights WT[N][K], K^T, V^T).
// Lane layout (ISA): n = lane&15, half = lane>>4, same K order as load_A_vec:
// 8 contiguous at K = 8*half, 8 contiguous at K = 16+8*half => two 16-byte loads.
__device__ __forceinline__ v16bf load_Bt_vec(const unsigned short* p, int ld, int dbase, int lane) {
    int n = lane & 15, half = lane >> 4;
    const uint4* q = (const uint4*)(p + (size_t)n * ld + dbase + half * 8);
    union { v16bf v; uint4 q[2]; } r;
    r.q[0] = q[0];
    r.q[1] = q[2];
    return r.v;
}

// ---------------- elementwise utility kernels ----------------
__global__ void f2bf_kernel(const float* __restrict__ in, unsigned short* __restrict__ out, size_t n) {
    size_t i = (size_t)blockIdx.x * 256 + threadIdx.x;
    if (i < n) out[i] = f32_to_bf16_bits(in[i]);
}

// W [K][N] f32 -> WT [N][K] bf16
__global__ void wtrans_kernel(const float* __restrict__ W, unsigned short* __restrict__ WT) {
    size_t gid = (size_t)blockIdx.x * 256 + threadIdx.x;
    if (gid >= (size_t)HID * HID) return;
    int k = gid % HID;
    int n = gid / HID;
    WT[gid] = f32_to_bf16_bits(W[(size_t)k * HID + n]);
}

__global__ void memset32_kernel(unsigned* __restrict__ p, size_t n) {
    size_t i = (size_t)blockIdx.x * 256 + threadIdx.x;
    if (i < n) p[i] = 0u;
}

// ---------------- WMMA GEMM: C[M,N] = A[M,K] @ WT[N,K]^T + bias (+resid) ----------------
// One wave computes a 16x64 tile: A fragment reused across 4 WMMAs per k-step.
template <bool HASRES>
__global__ __launch_bounds__(256)
void wmma_gemm_kernel(const unsigned short* __restrict__ A,
                      const unsigned short* __restrict__ BT,   // [N][K] bf16
                      const float* __restrict__ bias,
                      const float* __restrict__ resid,
                      float* __restrict__ C,
                      int M, int N, int K) {
    int lane = threadIdx.x & 31;
    int wave = threadIdx.x >> 5;                     // 8 waves / block
    int groups_n = N >> 6;                           // 64-wide column groups
    int tile = blockIdx.x * 8 + wave;
    int total = (M >> 4) * groups_n;
    if (tile >= total) return;                       // wave-uniform exit
    int tm = tile / groups_n, tg = tile % groups_n;

    const unsigned short* Ap = A + (size_t)tm * 16 * K;
    const unsigned short* B0 = BT + (size_t)(tg * 64) * K;

    v8f c0 = {}, c1 = {}, c2 = {}, c3 = {};
    for (int k = 0; k < K; k += 32) {
        v16bf a  = load_A_vec(Ap + k, K, lane);
        v16bf b0 = load_Bt_vec(B0,                    K, k, lane);
        v16bf b1 = load_Bt_vec(B0 + (size_t)16 * K,   K, k, lane);
        v16bf b2 = load_Bt_vec(B0 + (size_t)32 * K,   K, k, lane);
        v16bf b3 = load_Bt_vec(B0 + (size_t)48 * K,   K, k, lane);
        c0 = __builtin_amdgcn_wmma_f32_16x16x32_bf16(false, a, false, b0, (short)0, c0, false, false);
        c1 = __builtin_amdgcn_wmma_f32_16x16x32_bf16(false, a, false, b1, (short)0, c1, false, false);
        c2 = __builtin_amdgcn_wmma_f32_16x16x32_bf16(false, a, false, b2, (short)0, c2, false, false);
        c3 = __builtin_amdgcn_wmma_f32_16x16x32_bf16(false, a, false, b3, (short)0, c3, false, false);
    }

    int half = lane >> 4, nn = lane & 15;
    int colbase = tg * 64 + nn;
    float bv0 = bias[colbase +  0];
    float bv1 = bias[colbase + 16];
    float bv2 = bias[colbase + 32];
    float bv3 = bias[colbase + 48];
#pragma unroll
    for (int r = 0; r < 8; ++r) {
        size_t row = (size_t)(tm * 16 + r + 8 * half);
        size_t idx = row * N + colbase;
        float v0 = c0[r] + bv0;
        float v1 = c1[r] + bv1;
        float v2 = c2[r] + bv2;
        float v3 = c3[r] + bv3;
        if (HASRES) {
            v0 += resid[idx +  0];
            v1 += resid[idx + 16];
            v2 += resid[idx + 32];
            v3 += resid[idx + 48];
        }
        C[idx +  0] = v0;
        C[idx + 16] = v1;
        C[idx + 32] = v2;
        C[idx + 48] = v3;
    }
}

// ---------------- metric = mean over heads of k ----------------
__global__ void metric_mean_kernel(const float* __restrict__ k, float* __restrict__ metric) {
    size_t gid = (size_t)blockIdx.x * 256 + threadIdx.x;
    if (gid >= (size_t)BB * TT * DD) return;
    int d = gid % DD;
    size_t t = gid / DD;
    const float* kp = k + t * HID + d;
    float s = 0.f;
#pragma unroll
    for (int h = 0; h < HH; ++h) s += kp[h * DD];
    metric[gid] = s * (1.f / HH);
}

__global__ __launch_bounds__(64)
void metric_norm_kernel(float* __restrict__ metric) {
    size_t row = blockIdx.x;
    int t = threadIdx.x;
    float x = metric[row * DD + t];
    __shared__ float sm[64];
    sm[t] = x * x;
    __syncthreads();
    for (int s = 32; s > 0; s >>= 1) { if (t < s) sm[t] += sm[t + s]; __syncthreads(); }
    float nrm = sqrtf(sm[0]) + 1e-6f;
    metric[row * DD + t] = x / nrm;
}

// ---------------- per-even-token best match among odd tokens ----------------
__global__ __launch_bounds__(64)
void argmax_kernel(const float* __restrict__ metric, float* __restrict__ node_max, int* __restrict__ node_idx) {
    int i = blockIdx.x & 1023;
    int b = blockIdx.x >> 10;
    int t = threadIdx.x;
    __shared__ float av[64];
    __shared__ float bv[64];
    __shared__ int   bi[64];
    av[t] = metric[((size_t)b * TT + 2 * i) * DD + t];
    __syncthreads();
    float best = -1e30f; int bidx = 0;
    for (int j = t; j < TT / 2; j += 64) {
        const float* bp = metric + ((size_t)b * TT + 2 * j + 1) * DD;
        float dot = 0.f;
#pragma unroll 8
        for (int d = 0; d < DD; ++d) dot += av[d] * bp[d];
        if (dot > best) { best = dot; bidx = j; }
    }
    bv[t] = best; bi[t] = bidx;
    __syncthreads();
    for (int s = 32; s > 0; s >>= 1) {
        if (t < s) {
            if (bv[t + s] > bv[t] || (bv[t + s] == bv[t] && bi[t + s] < bi[t])) {
                bv[t] = bv[t + s]; bi[t] = bi[t + s];
            }
        }
        __syncthreads();
    }
    if (t == 0) { node_max[b * 1024 + i] = bv[0]; node_idx[b * 1024 + i] = bi[0]; }
}

// ---------------- bitonic argsort (descending) + index selection ----------------
__global__ __launch_bounds__(1024)
void sort_kernel(const float* __restrict__ node_max, const int* __restrict__ node_idx,
                 int* __restrict__ src_idx, int* __restrict__ dst_idx, int* __restrict__ unm_idx) {
    int b = blockIdx.x;
    int t = threadIdx.x;
    __shared__ float v[1024];
    __shared__ int   id[1024];
    v[t] = node_max[b * 1024 + t];
    id[t] = t;
    __syncthreads();
    for (int k = 2; k <= 1024; k <<= 1) {
        for (int j = k >> 1; j > 0; j >>= 1) {
            int ixj = t ^ j;
            if (ixj > t) {
                bool up = ((t & k) == 0);
                float vt = v[t], vx = v[ixj];
                bool sw = up ? (vt < vx) : (vt > vx);
                if (sw) {
                    v[t] = vx; v[ixj] = vt;
                    int tmp = id[t]; id[t] = id[ixj]; id[ixj] = tmp;
                }
            }
            __syncthreads();
        }
    }
    if (t < RR) {
        src_idx[b * RR + t] = id[t];
        dst_idx[b * RR + t] = node_idx[b * 1024 + id[t]];
    } else {
        unm_idx[b * RR + (t - RR)] = id[t];
    }
}

// ---------------- merge: counts / scatter-add / finalize ----------------
__global__ void count_kernel(const int* __restrict__ dst_idx, int* __restrict__ counts) {
    int gid = blockIdx.x * 256 + threadIdx.x;
    if (gid >= BB * RR) return;
    int b = gid / RR;
    atomicAdd(&counts[b * 1024 + dst_idx[gid]], 1);
}

__global__ void scatter_kernel(const float* __restrict__ X, const int* __restrict__ src_idx,
                               const int* __restrict__ dst_idx, float* __restrict__ sums) {
    size_t gid = (size_t)blockIdx.x * 256 + threadIdx.x;
    if (gid >= (size_t)BB * RR * HID) return;
    int c = gid % HID;
    size_t t = gid / HID;
    int r = t % RR;
    int b = t / RR;
    int s  = src_idx[b * RR + r];
    int dj = dst_idx[b * RR + r];
    atomicAdd(&sums[((size_t)b * 1024 + dj) * HID + c],
              X[((size_t)b * TT + 2 * s) * HID + c]);
}

// outf: f32 row-major [b][row][c]; outb: bf16 row-major; outbt: bf16 transposed [b][c][row]
__global__ void finalize_kernel(const float* __restrict__ X, const int* __restrict__ unm_idx,
                                const float* __restrict__ sums, const int* __restrict__ counts,
                                float* __restrict__ outf, unsigned short* __restrict__ outb,
                                unsigned short* __restrict__ outbt) {
    size_t gid = (size_t)blockIdx.x * 256 + threadIdx.x;
    if (gid >= (size_t)BB * T2 * HID) return;
    int c = gid % HID;
    size_t t = gid / HID;
    int row = t % T2;
    int b = t / T2;
    float val;
    if (row < RR) {
        int u = unm_idx[b * RR + row];
        val = X[((size_t)b * TT + 2 * u) * HID + c];
    } else {
        int j = row - RR;
        float cnt = (float)counts[b * 1024 + j];
        val = (X[((size_t)b * TT + 2 * j + 1) * HID + c] +
               sums[((size_t)b * 1024 + j) * HID + c]) / (1.f + cnt);
    }
    if (outf)  outf[gid] = val;
    if (outb)  outb[gid] = f32_to_bf16_bits(val);
    if (outbt) outbt[((size_t)b * HID + c) * T2 + row] = f32_to_bf16_bits(val);
}

// ---------------- flash attention: one wave per 16-query tile ----------------
// Q, K row-major bf16 [b][token][HID]; VT transposed bf16 [b][channel][T2].
__global__ __launch_bounds__(32)
void attn_kernel(const unsigned short* __restrict__ Q,
                 const unsigned short* __restrict__ K,
                 const unsigned short* __restrict__ VT,
                 unsigned short* __restrict__ ctx) {
    const int lane = threadIdx.x;
    const int half = lane >> 4;
    const int nn   = lane & 15;
    const int qt = blockIdx.x % (T2 / 16);
    const int bh = blockIdx.x / (T2 / 16);
    const int h = bh % HH;
    const int b = bh / HH;

    const unsigned short* Qp    = Q  + ((size_t)(b * T2 + qt * 16)) * HID + h * DD;
    const unsigned short* Kbase = K  + (size_t)b * T2 * HID + h * DD;
    const unsigned short* Vt    = VT + ((size_t)b * HID + h * DD) * T2;

    v16bf qa0 = load_A_vec(Qp, HID, lane);        // d 0..31
    v16bf qa1 = load_A_vec(Qp + 32, HID, lane);   // d 32..63

    __shared__ float P[16][32];

    v8f o0 = {}, o1 = {}, o2 = {}, o3 = {};
    float mrow[8], lrow[8];
#pragma unroll
    for (int r = 0; r < 8; ++r) { mrow[r] = -1e30f; lrow[r] = 0.f; }

    for (int kt = 0; kt < T2; kt += 32) {
        // S = Q·K^T for key tokens [kt, kt+32)
        const unsigned short* Kp0 = Kbase + (size_t)kt * HID;
        const unsigned short* Kp1 = Kbase + (size_t)(kt + 16) * HID;
        v8f s0 = {}, s1 = {};
        {
            v16bf kb0 = load_Bt_vec(Kp0, HID, 0, lane);
            v16bf kb1 = load_Bt_vec(Kp0, HID, 32, lane);
            s0 = __builtin_amdgcn_wmma_f32_16x16x32_bf16(false, qa0, false, kb0, (short)0, s0, false, false);
            s0 = __builtin_amdgcn_wmma_f32_16x16x32_bf16(false, qa1, false, kb1, (short)0, s0, false, false);
        }
        {
            v16bf kb0 = load_Bt_vec(Kp1, HID, 0, lane);
            v16bf kb1 = load_Bt_vec(Kp1, HID, 32, lane);
            s1 = __builtin_amdgcn_wmma_f32_16x16x32_bf16(false, qa0, false, kb0, (short)0, s1, false, false);
            s1 = __builtin_amdgcn_wmma_f32_16x16x32_bf16(false, qa1, false, kb1, (short)0, s1, false, false);
        }
        // online softmax; rows live in 16-lane halves -> shfl_xor 8/4/2/1
#pragma unroll
        for (int r = 0; r < 8; ++r) {
            float a = s0[r] * SCALE, c = s1[r] * SCALE;
            float mx = fmaxf(a, c);
            mx = fmaxf(mx, __shfl_xor(mx, 8, 32));
            mx = fmaxf(mx, __shfl_xor(mx, 4, 32));
            mx = fmaxf(mx, __shfl_xor(mx, 2, 32));
            mx = fmaxf(mx, __shfl_xor(mx, 1, 32));
            float mnew = fmaxf(mrow[r], mx);
            float corr = __expf(mrow[r] - mnew);
            float p0 = __expf(a - mnew);
            float p1 = __expf(c - mnew);
            float ps = p0 + p1;
            ps += __shfl_xor(ps, 8, 32);
            ps += __shfl_xor(ps, 4, 32);
            ps += __shfl_xor(ps, 2, 32);
            ps += __shfl_xor(ps, 1, 32);
            lrow[r] = lrow[r] * corr + ps;
            mrow[r] = mnew;
            o0[r] *= corr; o1[r] *= corr; o2[r] *= corr; o3[r] *= corr;
            P[r + 8 * half][nn]      = p0;
            P[r + 8 * half][16 + nn] = p1;
        }
        __syncthreads();
        // P (16x32) C-layout -> A-layout via LDS, convert to bf16
        v16bf pa;
#pragma unroll
        for (int i = 0; i < 8; ++i) {
            int kb = ((i < 4) ? (2 * i) : (16 + 2 * (i - 4))) + half * 8;
            pa[2 * i]     = us_to_bf(f32_to_bf16_bits(P[nn][kb]));
            pa[2 * i + 1] = us_to_bf(f32_to_bf16_bits(P[nn][kb + 1]));
        }
        __syncthreads();
        // O += P @ V : V^T rows are channels -> vectorized Bt loads
        v16bf vb0 = load_Bt_vec(Vt + (size_t) 0 * T2, T2, kt, lane);
        v16bf vb1 = load_Bt_vec(Vt + (size_t)16 * T2, T2, kt, lane);
        v16bf vb2 = load_Bt_vec(Vt + (size_t)32 * T2, T2, kt, lane);
        v16bf vb3 = load_Bt_vec(Vt + (size_t)48 * T2, T2, kt, lane);
        o0 = __builtin_amdgcn_wmma_f32_16x16x32_bf16(false, pa, false, vb0, (short)0, o0, false, false);
        o1 = __builtin_amdgcn_wmma_f32_16x16x32_bf16(false, pa, false, vb1, (short)0, o1, false, false);
        o2 = __builtin_amdgcn_wmma_f32_16x16x32_bf16(false, pa, false, vb2, (short)0, o2, false, false);
        o3 = __builtin_amdgcn_wmma_f32_16x16x32_bf16(false, pa, false, vb3, (short)0, o3, false, false);
    }

#pragma unroll
    for (int r = 0; r < 8; ++r) {
        float inv = 1.f / lrow[r];
        size_t base = ((size_t)(b * T2 + qt * 16 + r + 8 * half)) * HID + h * DD + nn;
        ctx[base + 0]  = f32_to_bf16_bits(o0[r] * inv);
        ctx[base + 16] = f32_to_bf16_bits(o1[r] * inv);
        ctx[base + 32] = f32_to_bf16_bits(o2[r] * inv);
        ctx[base + 48] = f32_to_bf16_bits(o3[r] * inv);
    }
}

// ---------------- layernorm ----------------
__global__ __launch_bounds__(256)
void layernorm_kernel(const float* __restrict__ x, const float* __restrict__ gamma,
                      const float* __restrict__ beta, float* __restrict__ out) {
    int row = blockIdx.x;
    int t = threadIdx.x;
    const float* xp = x + (size_t)row * HID;
    float s = 0.f, s2 = 0.f;
    for (int c = t; c < HID; c += 256) { float v = xp[c]; s += v; s2 += v * v; }
    __shared__ float sm[256], sm2[256];
    sm[t] = s; sm2[t] = s2;
    __syncthreads();
    for (int k = 128; k > 0; k >>= 1) {
        if (t < k) { sm[t] += sm[t + k]; sm2[t] += sm2[t + k]; }
        __syncthreads();
    }
    float mu  = sm[0] * (1.f / HID);
    float var = sm2[0] * (1.f / HID) - mu * mu;
    float inv = rsqrtf(var + 1e-12f);
    for (int c = t; c < HID; c += 256)
        out[(size_t)row * HID + c] = (xp[c] - mu) * inv * gamma[c] + beta[c];
}

// ---------------- host launcher ----------------
extern "C" void kernel_launch(void* const* d_in, const int* in_sizes, int n_in,
                              void* d_out, int out_size, void* d_ws, size_t ws_size,
                              hipStream_t stream) {
    (void)in_sizes; (void)n_in; (void)out_size; (void)ws_size;
    const float* hs    = (const float*)d_in[0];
    const float* Wq    = (const float*)d_in[1];
    const float* bq    = (const float*)d_in[2];
    const float* Wk    = (const float*)d_in[3];
    const float* bk    = (const float*)d_in[4];
    const float* Wv    = (const float*)d_in[5];
    const float* bv    = (const float*)d_in[6];
    const float* Wo    = (const float*)d_in[7];
    const float* bo    = (const float*)d_in[8];
    const float* gamma = (const float*)d_in[9];
    const float* beta  = (const float*)d_in[10];
    float* out = (float*)d_out;

    // ---- bump allocator over workspace ----
    char* wp = (char*)d_ws;
    auto alloc = [&](size_t bytes) -> void* {
        void* r = (void*)wp;
        wp += (bytes + 255) & ~(size_t)255;
        return r;
    };
    const size_t nHS  = (size_t)BB * TT * HID;
    const size_t nW   = (size_t)HID * HID;
    const size_t nMRG = (size_t)BB * T2 * HID;

    unsigned short* hsb  = (unsigned short*)alloc(nHS * 2);
    unsigned short* wqt  = (unsigned short*)alloc(nW * 2);    // transposed bf16 weights
    unsigned short* wkt  = (unsigned short*)alloc(nW * 2);
    unsigned short* wvt  = (unsigned short*)alloc(nW * 2);
    unsigned short* wot  = (unsigned short*)alloc(nW * 2);
    float* qf = (float*)alloc(nHS * 4);
    float* kf = (float*)alloc(nHS * 4);
    float* vf = (float*)alloc(nHS * 4);
    float* metric   = (float*)alloc((size_t)BB * TT * DD * 4);
    float* node_max = (float*)alloc((size_t)BB * 1024 * 4);
    int*   node_idx = (int*)alloc((size_t)BB * 1024 * 4);
    int*   src_idx  = (int*)alloc((size_t)BB * RR * 4);
    int*   dst_idx  = (int*)alloc((size_t)BB * RR * 4);
    int*   unm_idx  = (int*)alloc((size_t)BB * RR * 4);
    int*   counts   = (int*)alloc((size_t)BB * 1024 * 4);
    float* sums     = (float*)alloc((size_t)BB * 1024 * HID * 4);
    float* resm     = (float*)alloc(nMRG * 4);
    unsigned short* mqb  = (unsigned short*)alloc(nMRG * 2);
    unsigned short* mkb  = (unsigned short*)alloc(nMRG * 2);
    unsigned short* vtb  = (unsigned short*)alloc(nMRG * 2);  // merged V, transposed [b][c][t]
    unsigned short* ctxb = (unsigned short*)alloc(nMRG * 2);
    float* proj = (float*)alloc(nMRG * 4);

    auto blk = [](size_t n, int b) { return (unsigned)((n + b - 1) / b); };

    // 1) activations -> bf16; weights -> transposed bf16
    f2bf_kernel<<<blk(nHS, 256), 256, 0, stream>>>(hs, hsb, nHS);
    wtrans_kernel<<<blk(nW, 256), 256, 0, stream>>>(Wq, wqt);
    wtrans_kernel<<<blk(nW, 256), 256, 0, stream>>>(Wk, wkt);
    wtrans_kernel<<<blk(nW, 256), 256, 0, stream>>>(Wv, wvt);
    wtrans_kernel<<<blk(nW, 256), 256, 0, stream>>>(Wo, wot);

    // 2) QKV projections (WMMA)
    {
        int M = BB * TT, N = HID, K = HID;
        unsigned nblk = blk((size_t)(M / 16) * (N / 64), 8);
        wmma_gemm_kernel<false><<<nblk, 256, 0, stream>>>(hsb, wqt, bq, nullptr, qf, M, N, K);
        wmma_gemm_kernel<false><<<nblk, 256, 0, stream>>>(hsb, wkt, bk, nullptr, kf, M, N, K);
        wmma_gemm_kernel<false><<<nblk, 256, 0, stream>>>(hsb, wvt, bv, nullptr, vf, M, N, K);
    }

    // 3) merge indices
    metric_mean_kernel<<<blk((size_t)BB * TT * DD, 256), 256, 0, stream>>>(kf, metric);
    metric_norm_kernel<<<BB * TT, 64, 0, stream>>>(metric);
    argmax_kernel<<<BB * 1024, 64, 0, stream>>>(metric, node_max, node_idx);
    sort_kernel<<<BB, 1024, 0, stream>>>(node_max, node_idx, src_idx, dst_idx, unm_idx);

    // 4) counts
    memset32_kernel<<<blk((size_t)BB * 1024, 256), 256, 0, stream>>>((unsigned*)counts, (size_t)BB * 1024);
    count_kernel<<<blk((size_t)BB * RR, 256), 256, 0, stream>>>(dst_idx, counts);

    // 5) merge: q,k -> bf16 row-major; v -> bf16 transposed; residual -> f32
    const size_t nSums = (size_t)BB * 1024 * HID;
    const size_t nScat = (size_t)BB * RR * HID;
    const float* srcs[4] = { qf, kf, vf, hs };
    unsigned short* outsb[4]  = { mqb, mkb, nullptr, nullptr };
    unsigned short* outsbt[4] = { nullptr, nullptr, vtb, nullptr };
    float* outsf[4] = { nullptr, nullptr, nullptr, resm };
    for (int x = 0; x < 4; ++x) {
        memset32_kernel<<<blk(nSums, 256), 256, 0, stream>>>((unsigned*)sums, nSums);
        scatter_kernel<<<blk(nScat, 256), 256, 0, stream>>>(srcs[x], src_idx, dst_idx, sums);
        finalize_kernel<<<blk(nMRG, 256), 256, 0, stream>>>(srcs[x], unm_idx, sums, counts,
                                                            outsf[x], outsb[x], outsbt[x]);
    }

    // 6) flash attention (WMMA, 1 wave / 16-query tile)
    attn_kernel<<<BB * HH * (T2 / 16), 32, 0, stream>>>(mqb, mkb, vtb, ctxb);

    // 7) output projection + bias + residual (WMMA)
    {
        int M = BB * T2, N = HID, K = HID;
        unsigned nblk = blk((size_t)(M / 16) * (N / 64), 8);
        wmma_gemm_kernel<true><<<nblk, 256, 0, stream>>>(ctxb, wot, bo, resm, proj, M, N, K);
    }

    // 8) layernorm -> d_out
    layernorm_kernel<<<BB * T2, 256, 0, stream>>>(proj, gamma, beta, out);
}
